// spatial_attention_88845693485343
// MI455X (gfx1250) — matrix operand order repaired and checked by reading, not compiled, not run on hardware
//
#include <hip/hip_runtime.h>
#include <hip/hip_bf16.h>

typedef __attribute__((ext_vector_type(16))) _Float16 v16h;
typedef __attribute__((ext_vector_type(8)))  _Float16 v8h;
typedef __attribute__((ext_vector_type(4)))  _Float16 v4h;
typedef __attribute__((ext_vector_type(8)))  float    v8f;
typedef __attribute__((ext_vector_type(4)))  float    v4f;

#define BB 256
#define NN 256
#define DD 256
#define NHEAD 72
#define NBEAR 72

__constant__ const float kEPS = 1e-14f;

// ---------------------------------------------------------------------------
// Kernel 1: weights = masked softmax(relu(domain[idx1,idx2] - dist)) -> f16
// Block = 256 threads = 4 rows x 64 lanes; each lane handles 4 columns via
// float4 loads (b128 streaming reads of the three (B,N,N) tensors).
// ---------------------------------------------------------------------------
__global__ void k_weights(const float* __restrict__ dist,
                          const float* __restrict__ bear,
                          const float* __restrict__ head,
                          const float* __restrict__ mask,
                          const float* __restrict__ domain,
                          _Float16* __restrict__ wout) {
    const int b  = blockIdx.y;
    const int t  = threadIdx.x;
    const int r  = t >> 6;                    // row-in-block 0..3
    const int q  = t & 63;                    // lane group 0..63
    const int i  = blockIdx.x * 4 + r;        // row in N
    const int j0 = q * 4;                     // first of 4 columns
    const size_t base = ((size_t)b * NN + (size_t)i) * NN + j0;

    const v4f h4  = *(const v4f*)(head + base);
    const v4f br4 = *(const v4f*)(bear + base);
    const v4f d4  = *(const v4f*)(dist + base);
    const v4f mj4 = *(const v4f*)(mask + (size_t)b * NN + j0);
    const float mi = mask[(size_t)b * NN + i];

    float e[4];
    bool  valid[4];
    float partial = 0.0f;
    #pragma unroll
    for (int u = 0; u < 4; ++u) {
        int idx1 = (int)floorf((h4[u]  + 2.5f) * 0.2f);
        int idx2 = (int)floorf((br4[u] + 2.5f) * 0.2f);
        idx1 = min(max(idx1, 0), NHEAD - 1);
        idx2 = min(max(idx2, 0), NBEAR - 1);
        const float dom = domain[idx1 * NBEAR + idx2];
        float w = fmaxf(dom - d4[u], 0.0f);
        valid[u] = ((j0 + u) != i) && (mi * mj4[u] > 0.0f);
        w = valid[u] ? w : 0.0f;
        e[u] = ((w != 0.0f) ? __expf(w) : 0.0f) + kEPS;
        partial += e[u];
    }

    __shared__ float sred[4][64];
    sred[r][q] = partial;
    __syncthreads();
    #pragma unroll
    for (int s = 32; s > 0; s >>= 1) {
        if (q < s) sred[r][q] += sred[r][q + s];
        __syncthreads();
    }
    const float denom = sred[r][0] + kEPS;
    const float rden  = 1.0f / denom;

    v4h o;
    #pragma unroll
    for (int u = 0; u < 4; ++u) {
        o[u] = (_Float16)(valid[u] ? e[u] * rden : 0.0f);
    }
    *(v4h*)(wout + base) = o;
}

// ---------------------------------------------------------------------------
// Kernel 2: hidden f32 (B,N,D) -> h16 f16 (B,N,D)  AND  hT f16 (B,D,N).
// 32x33 LDS tile for the transposed store; both writes coalesced.
// ---------------------------------------------------------------------------
__global__ void k_convert_hidden(const float* __restrict__ hid,
                                 _Float16* __restrict__ h16,
                                 _Float16* __restrict__ hT) {
    __shared__ _Float16 tile[32][33];
    const int b  = blockIdx.x;
    const int i0 = blockIdx.y * 32;   // N offset
    const int j0 = blockIdx.z * 32;   // D offset
    const int t  = threadIdx.x;
    const int c  = t & 31;
    const int r  = t >> 5;            // 0..7

    const float* src = hid + ((size_t)b * NN + i0) * DD + j0;
    _Float16*    ds  = h16 + ((size_t)b * NN + i0) * DD + j0;
    #pragma unroll
    for (int rr = 0; rr < 4; ++rr) {
        const int row = r + rr * 8;
        const _Float16 v = (_Float16)src[(size_t)row * DD + c];
        ds[(size_t)row * DD + c] = v;
        tile[row][c] = v;
    }
    __syncthreads();
    _Float16* dt = hT + ((size_t)b * DD + j0) * NN + i0;
    #pragma unroll
    for (int rr = 0; rr < 4; ++rr) {
        const int row = r + rr * 8;
        dt[(size_t)row * NN + c] = tile[c][row];
    }
}

// ---------------------------------------------------------------------------
// Kernel 3: W (D, 2D) f32 -> f16 (one-shot, 131072 elements)
// ---------------------------------------------------------------------------
__global__ void k_convert_w(const float* __restrict__ Wm,
                            _Float16* __restrict__ Wf) {
    const size_t idx = ((size_t)blockIdx.x * 256 + threadIdx.x) * 4;
    const v4f v = *(const v4f*)(Wm + idx);
    v4h o;
    #pragma unroll
    for (int u = 0; u < 4; ++u) o[u] = (_Float16)v[u];
    *(v4h*)(Wf + idx) = o;
}

// Build a 16-bit A fragment (16x32) for row-major f16 A with leading dim lda.
// Lane L<16 : a[0..15]=A[L][k..k+15]
// Lane L>=16: a[0..15]=A[L][k+16..k+31]
__device__ __forceinline__ v16h load_a_f16(const _Float16* __restrict__ A,
                                           int row, int lda, int k, int sel) {
    const _Float16* pa = A + (size_t)row * lda + k + sel * 16;
    const v8h a0 = *(const v8h*)pa;
    const v8h a1 = *(const v8h*)(pa + 8);
    v16h a;
    #pragma unroll
    for (int qq = 0; qq < 8; ++qq) { a[qq] = a0[qq]; a[qq + 8] = a1[qq]; }
    return a;
}

// ---------------------------------------------------------------------------
// Kernel 4: weighted[b] = weights[b] (NxN,f16) @ hidden[b] (NxD via hT,f16)
// One wave -> 16x64 tile (4 accumulators); block = 4 waves -> 64x64 tile.
// ---------------------------------------------------------------------------
__global__ void k_gemm1(const _Float16* __restrict__ Wgt,  // (B,N,N)
                        const _Float16* __restrict__ hT,   // (B,D,N)
                        _Float16* __restrict__ WH) {       // (B,N,D)
    const int b    = blockIdx.x;
    const int lane = threadIdx.x & 31;
    const int wave = threadIdx.x >> 5;
    const int row0 = blockIdx.y * 64 + wave * 16;
    const int col0 = blockIdx.z * 64;
    const int lr   = lane & 15;
    const int sel  = lane >> 4;

    const _Float16* A  = Wgt + (size_t)b * NN * NN;
    const _Float16* Bh = hT  + (size_t)b * DD * NN;

    v8f acc[4] = {};
    for (int k = 0; k < NN; k += 32) {
        const v16h a = load_a_f16(A, row0 + lr, NN, k, sel);
        #pragma unroll
        for (int tt = 0; tt < 4; ++tt) {
            const int c0 = col0 + tt * 16;
            // lane n holds B[k+sel*16 + 0..15][c0+n] = hT[c0+n][k+sel*16 ...]
            const v16h bf = *(const v16h*)(Bh + (size_t)(c0 + lr) * NN + k + sel * 16);
            acc[tt] = __builtin_amdgcn_wmma_f32_16x16x32_f16(
                false, a, false, bf, (short)0, acc[tt], false, false);
        }
    }
    _Float16* O = WH + (size_t)b * NN * DD;
    #pragma unroll
    for (int tt = 0; tt < 4; ++tt) {
        const int c0 = col0 + tt * 16;
        #pragma unroll
        for (int rr = 0; rr < 8; ++rr) {
            O[(size_t)(row0 + sel * 8 + rr) * DD + c0 + lr] = (_Float16)acc[tt][rr];
        }
    }
}

// ---------------------------------------------------------------------------
// Kernel 5: out = [weighted | hidden] @ W^T + bias   (f32 out)
// All-f16 fragments: A from WH then h16; B[k][d] = Wf[d][k] -> per-lane
// contiguous 32B v16h loads of Wf rows.
// ---------------------------------------------------------------------------
__global__ void k_gemm2(const _Float16* __restrict__ WH,   // (B,N,D) f16
                        const _Float16* __restrict__ h16,  // (B,N,D) f16
                        const _Float16* __restrict__ Wf,   // (D, 2D) f16
                        const float* __restrict__ bias,    // (D,)
                        float* __restrict__ out) {         // (B,N,D) f32
    const int b    = blockIdx.x;
    const int lane = threadIdx.x & 31;
    const int wave = threadIdx.x >> 5;
    const int row0 = blockIdx.y * 64 + wave * 16;
    const int col0 = blockIdx.z * 64;
    const int lr   = lane & 15;
    const int sel  = lane >> 4;
    const int ldw  = 2 * DD;

    v8f acc[4] = {};

    // ---- phase 1: weighted @ W1^T ----
    const _Float16* A1 = WH + (size_t)b * NN * DD;
    for (int k = 0; k < DD; k += 32) {
        const v16h a = load_a_f16(A1, row0 + lr, DD, k, sel);
        #pragma unroll
        for (int tt = 0; tt < 4; ++tt) {
            const v16h bf = *(const v16h*)(Wf + (size_t)(col0 + tt * 16 + lr) * ldw + k + sel * 16);
            acc[tt] = __builtin_amdgcn_wmma_f32_16x16x32_f16(
                false, a, false, bf, (short)0, acc[tt], false, false);
        }
    }

    // ---- phase 2: hidden @ W2^T ----
    const _Float16* A2 = h16 + (size_t)b * NN * DD;
    for (int k = 0; k < DD; k += 32) {
        const v16h a = load_a_f16(A2, row0 + lr, DD, k, sel);
        #pragma unroll
        for (int tt = 0; tt < 4; ++tt) {
            const v16h bf = *(const v16h*)(Wf + (size_t)(col0 + tt * 16 + lr) * ldw + DD + k + sel * 16);
            acc[tt] = __builtin_amdgcn_wmma_f32_16x16x32_f16(
                false, a, false, bf, (short)0, acc[tt], false, false);
        }
    }

    // ---- bias + store ----
    float* O = out + (size_t)b * NN * DD;
    #pragma unroll
    for (int tt = 0; tt < 4; ++tt) {
        const int c0 = col0 + tt * 16;
        const float bv = bias[c0 + lr];
        #pragma unroll
        for (int rr = 0; rr < 8; ++rr) {
            O[(size_t)(row0 + sel * 8 + rr) * DD + c0 + lr] = acc[tt][rr] + bv;
        }
    }
}

// ---------------------------------------------------------------------------
extern "C" void kernel_launch(void* const* d_in, const int* in_sizes, int n_in,
                              void* d_out, int out_size, void* d_ws, size_t ws_size,
                              hipStream_t stream) {
    const float* hidden = (const float*)d_in[0];   // (B,N,D)
    const float* dist   = (const float*)d_in[1];   // (B,N,N)
    const float* bear   = (const float*)d_in[2];   // (B,N,N)
    const float* head   = (const float*)d_in[3];   // (B,N,N)
    const float* mask   = (const float*)d_in[4];   // (B,N)
    const float* domain = (const float*)d_in[5];   // (72,72)
    const float* Wm     = (const float*)d_in[6];   // (D,2D)
    const float* bias   = (const float*)d_in[7];   // (D,)
    float* out = (float*)d_out;

    // workspace: weights | hT | weighted | h16 | Wf   (all f16)
    _Float16* w16 = (_Float16*)d_ws;
    _Float16* hT  = w16 + (size_t)BB * NN * NN;
    _Float16* wh  = hT  + (size_t)BB * DD * NN;
    _Float16* h16 = wh  + (size_t)BB * NN * DD;
    _Float16* Wf  = h16 + (size_t)BB * NN * DD;
    (void)in_sizes; (void)n_in; (void)out_size; (void)ws_size;

    // 1) weights + masked softmax (4 rows per block, float4 streams)
    k_weights<<<dim3(NN / 4, BB), 256, 0, stream>>>(dist, bear, head, mask, domain, w16);

    // 2) hidden -> h16 (row-major) + hT (transposed), both f16
    k_convert_hidden<<<dim3(BB, NN / 32, DD / 32), 256, 0, stream>>>(hidden, h16, hT);

    // 3) W -> f16
    k_convert_w<<<dim3((DD * 2 * DD) / (256 * 4)), 256, 0, stream>>>(Wm, Wf);

    // 4) weighted = weights @ hidden   (f16 WMMA, f32 accum)
    k_gemm1<<<dim3(BB, NN / 64, NN / 64), 128, 0, stream>>>(w16, hT, wh);

    // 5) out = [weighted | hidden] @ W^T + bias
    k_gemm2<<<dim3(BB, NN / 64, DD / 64), 128, 0, stream>>>(wh, h16, Wf, bias, out);
}
